// G0_model_29162827939890
// MI455X (gfx1250) — matrix operand order repaired and checked by reading, not compiled, not run on hardware
//
#include <hip/hip_runtime.h>

// ---------------------------------------------------------------------------
// 2-layer LSTM (H=20, T=256, B=131072) + 20->64->1 ReLU head for MI455X.
// One wave32 owns 16 batch rows for the whole sequence. Gate GEMMs run on
// v_wmma_f32_16x16x32_f16 with gates padded to 32 per type (g = 32*type + h)
// so i/f/g/o of a given (b,h) are lane-local in the WMMA C layout. c-state
// stays in C layout in VGPRs; h round-trips through a wave-private LDS tile
// [b][h] (fp16) to become the next A operand. Weights live in B-operand
// VGPRs. Biases are folded into the activations (loop-invariant, hoisted),
// letting every WMMA use the inline-0 C operand (no accumulator-copy movs).
// ---------------------------------------------------------------------------

typedef __attribute__((ext_vector_type(16))) _Float16 v16h;
typedef __attribute__((ext_vector_type(8)))  _Float16 v8h;
typedef __attribute__((ext_vector_type(8)))  float    v8f;

#define HID   20     // true hidden size
#define HP    32     // padded hidden size (K dim)
#define SEQ_T 256
#define FFD   64
#define NT    8      // N tiles: 128 padded gate columns
#define BPW   16     // batch rows per wave
#define WPB   8      // waves per block
#define PITCH 40     // LDS row pitch in halfs (80B, 16B aligned)

// ---- transcendental helpers (native v_tanh_f32 confirmed on gfx1250) -------
__device__ __forceinline__ float dev_exp2(float x) { return __builtin_amdgcn_exp2f(x); }
__device__ __forceinline__ float dev_rcp (float x) { return __builtin_amdgcn_rcpf(x); }

#if __has_builtin(__builtin_amdgcn_tanhf)
#define NATIVE_TANH 1
__device__ __forceinline__ float dev_tanh(float x) { return __builtin_amdgcn_tanhf(x); }
#elif __has_builtin(__builtin_amdgcn_tanh_f32)
#define NATIVE_TANH 1
__device__ __forceinline__ float dev_tanh(float x) { return __builtin_amdgcn_tanh_f32(x); }
#else
#define NATIVE_TANH 0
#endif

// Activations with the bias folded in; the b-dependent terms are
// loop-invariant per tile and hoisted out of the timestep loop.
#if NATIVE_TANH
__device__ __forceinline__ float act_sig(float z, float b) {
  // sigmoid(z+b) = 0.5*tanh(0.5*z + 0.5*b) + 0.5           (1 trans)
  return __builtin_fmaf(0.5f, dev_tanh(__builtin_fmaf(0.5f, z, 0.5f * b)), 0.5f);
}
__device__ __forceinline__ float act_tanh(float z, float b) {
  return dev_tanh(z + b);                                   // (1 trans)
}
__device__ __forceinline__ float act_tanh0(float x) { return dev_tanh(x); }
#else
__device__ __forceinline__ float act_sig(float z, float b) {
  // sigmoid(z+b) = 1/(1+2^(-(z+b)*log2e))                  (exp2 + rcp)
  const float L = 1.44269504088896340736f;
  return dev_rcp(1.0f + dev_exp2(__builtin_fmaf(z, -L, -L * b)));
}
__device__ __forceinline__ float act_tanh(float z, float b) {
  return __builtin_fmaf(2.0f, act_sig(2.0f * z, 2.0f * b), -1.0f);
}
__device__ __forceinline__ float act_tanh0(float x) { return act_tanh(x, 0.f); }
#endif

__global__ __launch_bounds__(WPB * 32, 1) void
lstm2_wmma_kernel(const float* __restrict__ diag,
                  const float* __restrict__ Wih1, const float* __restrict__ Whh1,
                  const float* __restrict__ bih1, const float* __restrict__ bhh1,
                  const float* __restrict__ Wih2, const float* __restrict__ Whh2,
                  const float* __restrict__ bih2, const float* __restrict__ bhh2,
                  const float* __restrict__ W1,   const float* __restrict__ b1,
                  const float* __restrict__ W2,   const float* __restrict__ b2,
                  float* __restrict__ out, int batch)
{
  // wave-private h staging tiles: [wave][layer][batch row][h column]
  __shared__ __align__(16) _Float16 hmat[WPB][2][BPW][PITCH];

  const int lane = threadIdx.x & 31;
  const int wv   = threadIdx.x >> 5;
  const int hl   = lane >> 4;     // lane half (A: K block / C: batch high bit)
  const int ln   = lane & 15;
  const int k    = lane;          // unused; the packing loop's per-element k shadows it
  const long b0  = ((long)blockIdx.x * WPB + wv) * BPW;
  if (b0 >= batch) return;

  // ---- pack weights into WMMA B-operand registers (lane = N column) -------
  // B element e of tile t holds Wp[k(e,hl)][16t+ln] (table K order), padded gate g = 32*type+h.
  // Layer-1 K rows: 0..19 = W_hh1 columns, 20 = W_ih1 (the x column), rest 0.
  v16h B1[NT], B2i[NT], B2h[NT];
#pragma unroll
  for (int t = 0; t < NT; ++t) {
#pragma unroll
    for (int e = 0; e < 16; ++e) {
      const int g  = 16 * t + ln;
      const int k  = (e < 8) ? (8 * hl + e) : (16 + 8 * hl + (e - 8));
      const int ty = g >> 5;
      const int h  = g & 31;
      float w1 = 0.f, w2i = 0.f, w2h = 0.f;
      if (h < HID) {                       // lane-dependent: all ln for even t, ln < 4 for odd t
        const int row = ty * HID + h;      // original PyTorch gate row (i,f,g,o)
        if (k < HID) {
          w1  = Whh1[row * HID + k];
          w2i = Wih2[row * HID + k];
          w2h = Whh2[row * HID + k];
        } else if (k == HID) {
          w1 = Wih1[row];                  // x column of layer 1
        }
      }
      B1 [t][e] = (_Float16)w1;
      B2i[t][e] = (_Float16)w2i;
      B2h[t][e] = (_Float16)w2h;
    }
  }

  // ---- biases in C layout: value depends only on gate column (= lane) -----
  float bv1[NT], bv2[NT];
#pragma unroll
  for (int t = 0; t < NT; ++t) {
    const int g  = 16 * t + ln;
    const int ty = g >> 5;
    const int h  = g & 31;
    const bool live = (h < HID);
    const int row = ty * HID + (live ? h : 0);
    bv1[t] = live ? (bih1[row] + bhh1[row]) : 0.f;
    bv2[t] = live ? (bih2[row] + bhh2[row]) : 0.f;
  }

  // ---- states: c in C layout [parity p][reg r] -> (b = r+8*hl, h = 16p+ln)
  float c1[2][8], c2[2][8];
#pragma unroll
  for (int p = 0; p < 2; ++p)
#pragma unroll
    for (int r = 0; r < 8; ++r) { c1[p][r] = 0.f; c2[p][r] = 0.f; }

  {
    _Float16* hz = &hmat[wv][0][0][0];
    for (int i = lane; i < 2 * BPW * PITCH; i += 32) hz[i] = (_Float16)0.f;
  }
  __builtin_amdgcn_wave_barrier();

  const long brow = (b0 + ln < (long)batch) ? (b0 + ln) : (long)batch - 1;
  const float* xrow = diag + (size_t)brow * SEQ_T;

  const v8f zacc = {};   // inline-0 C operand for every WMMA

#pragma unroll 1
  for (int t = 0; t < SEQ_T; ++t) {
    const float x = xrow[t];

    // ---- A operands from LDS [b][h]: lane=batch, 2x ds_load_b128 each -----
    // A layout: lo lanes K 0..7 & 16..23, hi lanes K 8..15 & 24..31.
    v8h a1lo = *(const v8h*)&hmat[wv][0][ln][8 * hl];
    v8h a1hi = *(const v8h*)&hmat[wv][0][ln][16 + 8 * hl];
    v8h a2lo = *(const v8h*)&hmat[wv][1][ln][8 * hl];
    v8h a2hi = *(const v8h*)&hmat[wv][1][ln][16 + 8 * hl];
    v16h A1x, A2h;
#pragma unroll
    for (int j = 0; j < 8; ++j) {
      A1x[j] = a1lo[j]; A1x[8 + j] = a1hi[j];
      A2h[j] = a2lo[j]; A2h[8 + j] = a2hi[j];
    }
    if (hl == 0) A1x[12] = (_Float16)x;     // K=20 slot carries x_t

    // ---- layer 1: z = [x,h1] x W1, 8 WMMAs with inline-0 accumulator ------
    v8f z[NT];
#pragma unroll
    for (int n = 0; n < NT; ++n)
      z[n] = __builtin_amdgcn_wmma_f32_16x16x32_f16(
                 false, A1x, false, B1[n], (short)0, zacc, false, false);

    // ---- layer-1 cell update (lane-local: tiles p,2+p,4+p,6+p) ------------
#pragma unroll
    for (int p = 0; p < 2; ++p) {
#pragma unroll
      for (int r = 0; r < 8; ++r) {
        const float ig = act_sig (z[0 + p][r], bv1[0 + p]);
        const float fg = act_sig (z[2 + p][r], bv1[2 + p]);
        const float gg = act_tanh(z[4 + p][r], bv1[4 + p]);
        const float og = act_sig (z[6 + p][r], bv1[6 + p]);
        const float cn = __builtin_fmaf(fg, c1[p][r], ig * gg);
        c1[p][r] = cn;
        hmat[wv][0][r + 8 * hl][16 * p + ln] = (_Float16)(og * act_tanh0(cn));
      }
    }
    __builtin_amdgcn_wave_barrier();        // DS in-order within wave

    // ---- reload fresh h1 as A operand for layer 2 -------------------------
    v8h n1lo = *(const v8h*)&hmat[wv][0][ln][8 * hl];
    v8h n1hi = *(const v8h*)&hmat[wv][0][ln][16 + 8 * hl];
    v16h A1n;
#pragma unroll
    for (int j = 0; j < 8; ++j) { A1n[j] = n1lo[j]; A1n[8 + j] = n1hi[j]; }

    // ---- layer 2: z = h1_new x Wih2 + h2 x Whh2, 16 WMMAs -----------------
#pragma unroll
    for (int n = 0; n < NT; ++n) {
      v8f acc = __builtin_amdgcn_wmma_f32_16x16x32_f16(
                    false, A1n, false, B2i[n], (short)0, zacc, false, false);
      z[n]    = __builtin_amdgcn_wmma_f32_16x16x32_f16(
                    false, A2h, false, B2h[n], (short)0, acc, false, false);
    }

#pragma unroll
    for (int p = 0; p < 2; ++p) {
#pragma unroll
      for (int r = 0; r < 8; ++r) {
        const float ig = act_sig (z[0 + p][r], bv2[0 + p]);
        const float fg = act_sig (z[2 + p][r], bv2[2 + p]);
        const float gg = act_tanh(z[4 + p][r], bv2[4 + p]);
        const float og = act_sig (z[6 + p][r], bv2[6 + p]);
        const float cn = __builtin_fmaf(fg, c2[p][r], ig * gg);
        c2[p][r] = cn;
        hmat[wv][1][r + 8 * hl][16 * p + ln] = (_Float16)(og * act_tanh0(cn));
      }
    }
    __builtin_amdgcn_wave_barrier();
  }

  // ---- FF head: y2 = relu(W2 @ relu(W1 @ h2 + b1) + b2), once per element -
  __builtin_amdgcn_wave_barrier();
  if (hl == 0 && b0 + ln < (long)batch) {
    float h2f[HID];
#pragma unroll
    for (int h = 0; h < HID; ++h) h2f[h] = (float)hmat[wv][1][ln][h];
    float y2 = b2[0];
#pragma unroll 4
    for (int j = 0; j < FFD; ++j) {
      float y1 = b1[j];
#pragma unroll
      for (int h = 0; h < HID; ++h) y1 = __builtin_fmaf(W1[j * HID + h], h2f[h], y1);
      y1 = fmaxf(y1, 0.f);
      y2 = __builtin_fmaf(W2[j], y1, y2);
    }
    out[b0 + ln] = fmaxf(y2, 0.f);
  }
}

extern "C" void kernel_launch(void* const* d_in, const int* in_sizes, int n_in,
                              void* d_out, int out_size, void* d_ws, size_t ws_size,
                              hipStream_t stream) {
  (void)in_sizes; (void)n_in; (void)d_ws; (void)ws_size;
  const float* diag = (const float*)d_in[0];
  const float* Wih1 = (const float*)d_in[1];
  const float* Whh1 = (const float*)d_in[2];
  const float* bih1 = (const float*)d_in[3];
  const float* bhh1 = (const float*)d_in[4];
  const float* Wih2 = (const float*)d_in[5];
  const float* Whh2 = (const float*)d_in[6];
  const float* bih2 = (const float*)d_in[7];
  const float* bhh2 = (const float*)d_in[8];
  const float* W1   = (const float*)d_in[9];
  const float* b1   = (const float*)d_in[10];
  const float* W2   = (const float*)d_in[11];
  const float* b2   = (const float*)d_in[12];
  float* out = (float*)d_out;

  const int batch = out_size;                 // y2 is [B, 1]
  const int per_block = WPB * BPW;            // 128 batch rows per block
  dim3 grid((batch + per_block - 1) / per_block), block(WPB * 32);
  hipLaunchKernelGGL(lstm2_wmma_kernel, grid, block, 0, stream,
                     diag, Wih1, Whh1, bih1, bhh1, Wih2, Whh2, bih2, bhh2,
                     W1, b1, W2, b2, out, batch);
}
